// Network_90975997264671
// MI455X (gfx1250) — compile-verified, matrix operand order checked by reading, not run on hardware
//
#include <hip/hip_runtime.h>

// ---------------------------------------------------------------------------
// Types / helpers
// ---------------------------------------------------------------------------
typedef __attribute__((ext_vector_type(16))) _Float16 v16h;
typedef __attribute__((ext_vector_type(8)))  _Float16 v8h;
typedef __attribute__((ext_vector_type(8)))  float    v8f;
typedef __attribute__((ext_vector_type(4)))  int      v4i;

union Vh16 { v16h v; v8h h[2]; _Float16 e[16]; };
union Vf8  { v8f  v; float e[8]; };

#define KSZ   51
#define KHALF 25

#if __has_builtin(__builtin_amdgcn_global_load_async_to_lds_b128) && \
    __has_builtin(__builtin_amdgcn_s_wait_asynccnt)
#define USE_ASYNC_LDS 1
#define GLOBAL_AS __attribute__((address_space(1)))
#define LDS_AS    __attribute__((address_space(3)))
typedef GLOBAL_AS v4i* gp_v4i;     // pointer to b128 in global memory
typedef LDS_AS    v4i* lp_v4i;     // pointer to b128 in LDS
#endif

static inline int icdiv(int a, int b) { return (a + b - 1) / b; }
static inline int pad32i(int c)       { return (c + 31) & ~31; }

// ---------------------------------------------------------------------------
// Weight repack:  OIHW f32  ->  WMMA-fragment-ordered f16
//   layout: [tap(9)][kChunk(CinPad/32)][grp(2)][cout(CoutPad)][16]
//   For lane-group g, fragment slot s maps to k = (s<8) ? g*8+s : 16+g*8+(s-8)
//   so each lane's whole 16-element B fragment is one contiguous 32B load.
// ---------------------------------------------------------------------------
__global__ __launch_bounds__(256) void repack_w_kernel(
    const float* __restrict__ w, _Float16* __restrict__ out,
    int Cin, int Cout, int CinPad, int CoutPad)
{
    const int nkc = CinPad >> 5;
    int idx = blockIdx.x * blockDim.x + threadIdx.x;
    int total = 9 * nkc * 2 * CoutPad * 16;
    if (idx >= total) return;
    int s  = idx & 15;
    int t  = idx >> 4;
    int co = t % CoutPad;  t /= CoutPad;
    int g  = t & 1;        t >>= 1;
    int kc = t % nkc;
    int tap = t / nkc;
    int k   = (s < 8) ? (g * 8 + s) : (16 + g * 8 + (s - 8));
    int cin = kc * 32 + k;
    float v = 0.f;
    if (cin < Cin && co < Cout)
        v = w[((size_t)co * Cin + cin) * 9 + tap];      // OIHW, tap = ky*3+kx
    out[idx] = (_Float16)v;
}

__global__ __launch_bounds__(64) void repack_b_kernel(
    const float* __restrict__ b, float* __restrict__ out, int Cout, int CoutPad)
{
    int idx = blockIdx.x * blockDim.x + threadIdx.x;
    if (idx >= CoutPad) return;
    out[idx] = (idx < Cout) ? b[idx] : 0.f;
}

// ---------------------------------------------------------------------------
// Zero the 1-pixel halo border of an NHWC-with-halo tensor [B][H+2][W+2][C]
// ---------------------------------------------------------------------------
__global__ __launch_bounds__(256) void zero_halo_kernel(
    _Float16* __restrict__ buf, int Bn, int H, int W, int C)
{
    const int Hp = H + 2, Wp = W + 2;
    int idx = blockIdx.x * blockDim.x + threadIdx.x;
    int total = Bn * Hp * Wp;
    if (idx >= total) return;
    int x = idx % Wp;
    int y = (idx / Wp) % Hp;
    if (x == 0 || x == Wp - 1 || y == 0 || y == Hp - 1) {
        unsigned long long* q = (unsigned long long*)(buf + (size_t)idx * C);
        for (int c = 0; c < (C >> 2); ++c) q[c] = 0ull;
    }
}

// ---------------------------------------------------------------------------
// Input pack: two f32 NCHW [B,3,H,W] -> f16 halo NHWC [B,H+2,W+2,32]
// ---------------------------------------------------------------------------
__global__ __launch_bounds__(256) void pack_kernel(
    const float* __restrict__ t1, const float* __restrict__ t2,
    _Float16* __restrict__ out, int Bn, int H, int W)
{
    int idx = blockIdx.x * blockDim.x + threadIdx.x;
    int total = Bn * H * W;
    if (idx >= total) return;
    int w = idx % W;
    int t = idx / W;
    int h = t % H;
    int b = t / H;
    _Float16* o = out + ((size_t)(b * (H + 2) + h + 1) * (W + 2) + w + 1) * 32;
    #pragma unroll
    for (int c = 0; c < 3; ++c) {
        o[c]     = (_Float16)t1[((size_t)(b * 3 + c) * H + h) * W + w];
        o[3 + c] = (_Float16)t2[((size_t)(b * 3 + c) * H + h) * W + w];
    }
    #pragma unroll
    for (int c = 6; c < 32; ++c) o[c] = (_Float16)0.f;
}

// ---------------------------------------------------------------------------
// Implicit-GEMM 3x3 SAME conv via WMMA f16->f32.
//   in  : halo [B][H+2][W+2][CinPad]
//   wf  : fragment-ordered weights (see repack_w_kernel)
//   out : halo [B][H+2][W+2][CoutPad] (interior written)
//   Block = 128 threads (4 waves): 64 spatial x 32 cout per block.
//   Wave  = 16 spatial (its quarter of the shared LDS A tile) x 32 cout
//           -> 2 WMMAs per K-chunk sharing one A fragment.
//   A tile double-buffered in LDS; staged with gfx1250 async-to-LDS DMA:
//   chunk c+1 is issued before waiting on chunk c (asynccnt<=2), so the DMA
//   overlaps the WMMAs of the current chunk.
// ---------------------------------------------------------------------------
__global__ __launch_bounds__(128) void conv3x3_wmma_kernel(
    const _Float16* __restrict__ in,
    const _Float16* __restrict__ wf,
    const float*    __restrict__ bias,
    _Float16*       __restrict__ out,
    int H, int W, int CinPad, int CoutPad, int relu)
{
    __shared__ _Float16 ldsA[2][64][40];  // ping-pong: 64 pos x 32 cins (+pad)

    const int tid  = threadIdx.x;
    const int lane = tid & 31;
    const int wave = tid >> 5;
    const int posBase  = blockIdx.x * 64;
    const int coutBase = blockIdx.y * 32;
    const int nkc = CinPad >> 5;
    const int Hp = H + 2, Wp = W + 2;
    const int hw = H * W;

    // cooperative A-tile loader: thread -> (position p, 16-f16 half)
    const int p    = tid >> 1;
    const int coff = (tid & 1) * 16;
    const int pos = posBase + p;
    const int b   = pos / hw;
    const int rem = pos - b * hw;
    const int ph  = rem / W;
    const int pw  = rem - ph * W;
    // tap (ky,kx): halo pixel (ph+ky, pw+kx); tap (0,0) base:
    const _Float16* pixBase =
        in + ((size_t)(b * Hp + ph) * Wp + pw) * CinPad + coff;

    // fragment lane mapping
    const int g    = lane >> 4;        // lane group (K-half select)
    const int kh   = g * 8;
    const int n    = lane & 15;        // cout column within tile
    const int mRow = wave * 16 + (lane & 15);   // A-tile row for this lane

    // stage K-chunk `cidx` of the A tile into LDS buffer `bufi`
    auto stage = [&](int cidx, int bufi) {
        const int tap_ = cidx / nkc;
        const int kc_  = cidx - tap_ * nkc;
        const int ky_  = tap_ / 3, kx_ = tap_ - 3 * ky_;
        const _Float16* sg =
            pixBase + ((size_t)ky_ * Wp + kx_) * CinPad + (size_t)kc_ * 32;
        _Float16* sl = &ldsA[bufi][p][coff];
#ifdef USE_ASYNC_LDS
        __builtin_amdgcn_global_load_async_to_lds_b128(
            (gp_v4i)sg, (lp_v4i)sl, 0, 0);
        __builtin_amdgcn_global_load_async_to_lds_b128(
            (gp_v4i)sg, (lp_v4i)sl, 16, 0);
#else
        v8h t0 = *(const v8h*)sg;
        v8h t1 = *(const v8h*)(sg + 8);
        *(v8h*)sl       = t0;
        *(v8h*)(sl + 8) = t1;
#endif
    };

    Vf8 acc0, acc1;
    #pragma unroll
    for (int r = 0; r < 8; ++r) { acc0.e[r] = 0.f; acc1.e[r] = 0.f; }

    const int nChunks = 9 * nkc;          // K-chunks: 9 taps x CinPad/32
    stage(0, 0);                          // prologue

    for (int c = 0; c < nChunks; ++c) {
        const int cur = c & 1;
        __syncthreads();                  // readers of buf[cur^1] (chunk c-1) done
        if (c + 1 < nChunks)
            stage(c + 1, cur ^ 1);        // overlap next chunk's DMA with math
#ifdef USE_ASYNC_LDS
        if (c + 1 < nChunks)
            __builtin_amdgcn_s_wait_asynccnt(2);  // older pair (chunk c) landed
        else
            __builtin_amdgcn_s_wait_asynccnt(0);
#endif
        __syncthreads();                  // chunk c visible to all waves

        // A fragment: two contiguous 16B LDS reads (ISA 7.12.2 layout)
        Vh16 a;
        a.h[0] = *(const v8h*)&ldsA[cur][mRow][kh];
        a.h[1] = *(const v8h*)&ldsA[cur][mRow][16 + kh];

        // B fragments: one contiguous 32B global load each (pre-permuted);
        // weight chunk index == c since (tap*nkc + kc) == c
        const _Float16* wp =
            wf + (((size_t)c * 2 + g) * CoutPad + coutBase) * 16;
        v16h b0 = *(const v16h*)(wp + (size_t)n * 16);
        v16h b1 = *(const v16h*)(wp + (size_t)(n + 16) * 16);
        __builtin_prefetch(wp + (size_t)2 * CoutPad * 16, 0, 0);

        acc0.v = __builtin_amdgcn_wmma_f32_16x16x32_f16(
                     false, a.v, false, b0, (short)0, acc0.v, false, false);
        acc1.v = __builtin_amdgcn_wmma_f32_16x16x32_f16(
                     false, a.v, false, b1, (short)0, acc1.v, false, false);
    }

    // epilogue: bias + ReLU + f16 store to halo interior
    const float bv0 = bias[coutBase + n];
    const float bv1 = bias[coutBase + 16 + n];
    #pragma unroll
    for (int r = 0; r < 8; ++r) {
        const int mm   = r + kh;                     // C/D layout M index
        const int pos2 = posBase + wave * 16 + mm;
        const int b2   = pos2 / hw;
        const int rm2  = pos2 - b2 * hw;
        const int h2   = rm2 / W;
        const int w2   = rm2 - h2 * W;
        const size_t oa = ((size_t)(b2 * Hp + h2 + 1) * Wp + (w2 + 1)) * CoutPad
                        + coutBase + n;
        float v0 = acc0.e[r] + bv0;
        float v1 = acc1.e[r] + bv1;
        if (relu) { v0 = fmaxf(v0, 0.f); v1 = fmaxf(v1, 0.f); }
        out[oa]      = (_Float16)v0;
        out[oa + 16] = (_Float16)v1;
    }
}

// ---------------------------------------------------------------------------
// 2x2 average pool, halo NHWC f16 -> halo NHWC f16
// ---------------------------------------------------------------------------
__global__ __launch_bounds__(256) void avgpool_kernel(
    const _Float16* __restrict__ in, _Float16* __restrict__ out,
    int Bn, int H, int W, int C)
{
    const int Ho = H / 2, Wo = W / 2;
    int idx = blockIdx.x * blockDim.x + threadIdx.x;
    int total = Bn * Ho * Wo * C;
    if (idx >= total) return;
    int c = idx % C;
    int t = idx / C;
    int j = t % Wo; t /= Wo;
    int i = t % Ho;
    int b = t / Ho;
    const size_t r0 = ((size_t)(b * (H + 2) + 2 * i + 1) * (W + 2) + 2 * j + 1) * C + c;
    const size_t r1 = ((size_t)(b * (H + 2) + 2 * i + 2) * (W + 2) + 2 * j + 1) * C + c;
    float s = (float)in[r0] + (float)in[r0 + C] +
              (float)in[r1] + (float)in[r1 + C];
    out[((size_t)(b * (Ho + 2) + i + 1) * (Wo + 2) + j + 1) * C + c] =
        (_Float16)(s * 0.25f);
}

// ---------------------------------------------------------------------------
// Bilinear x2 upsample, align_corners=True, halo NHWC f16
//   pos = arange(2n) * (n-1)/(2n-1)   (matches reference)
// ---------------------------------------------------------------------------
__global__ __launch_bounds__(256) void up2_kernel(
    const _Float16* __restrict__ in, _Float16* __restrict__ out,
    int Bn, int H, int W, int C)
{
    const int Ho = 2 * H, Wo = 2 * W;
    int idx = blockIdx.x * blockDim.x + threadIdx.x;
    int total = Bn * Ho * Wo * C;
    if (idx >= total) return;
    int c = idx % C;
    int t = idx / C;
    int j = t % Wo; t /= Wo;
    int i = t % Ho;
    int b = t / Ho;

    float py = (float)i * ((float)(H - 1) / (float)(Ho - 1));
    float px = (float)j * ((float)(W - 1) / (float)(Wo - 1));
    int y0 = (int)floorf(py); int y1 = min(y0 + 1, H - 1);
    int x0 = (int)floorf(px); int x1 = min(x0 + 1, W - 1);
    float ty = py - (float)y0, tx = px - (float)x0;

    const int Wp = W + 2;
    const size_t b00 = ((size_t)(b * (H + 2) + y0 + 1) * Wp + x0 + 1) * C + c;
    const size_t b01 = ((size_t)(b * (H + 2) + y0 + 1) * Wp + x1 + 1) * C + c;
    const size_t b10 = ((size_t)(b * (H + 2) + y1 + 1) * Wp + x0 + 1) * C + c;
    const size_t b11 = ((size_t)(b * (H + 2) + y1 + 1) * Wp + x1 + 1) * C + c;
    float v = (1.f - ty) * ((1.f - tx) * (float)in[b00] + tx * (float)in[b01])
            +        ty  * ((1.f - tx) * (float)in[b10] + tx * (float)in[b11]);
    out[((size_t)(b * (Ho + 2) + i + 1) * (Wo + 2) + j + 1) * C + c] = (_Float16)v;
}

// ---------------------------------------------------------------------------
// Elementwise add over the FULL halo extent (halo: 0+0 stays 0)
// ---------------------------------------------------------------------------
__global__ __launch_bounds__(256) void add_kernel(
    const _Float16* __restrict__ a, const _Float16* __restrict__ b,
    _Float16* __restrict__ out, int n)
{
    int idx = blockIdx.x * blockDim.x + threadIdx.x;
    if (idx >= n) return;
    out[idx] = (_Float16)((float)a[idx] + (float)b[idx]);
}

// ---------------------------------------------------------------------------
// Adaptive separable conv readout (K=51, edge-clamped image pad).
//   Coefficient tensors in halo NHWC layout [B][H+2][W+2][64] f16.
// ---------------------------------------------------------------------------
__global__ __launch_bounds__(256) void sepconv_kernel(
    const float* __restrict__ t1, const float* __restrict__ t2,
    const _Float16* __restrict__ V1,  const _Float16* __restrict__ V2,
    const _Float16* __restrict__ Hk1, const _Float16* __restrict__ Hk2,
    float* __restrict__ out, int Bn, int H, int W)
{
    int idx = blockIdx.x * blockDim.x + threadIdx.x;
    int total = Bn * 3 * H * W;
    if (idx >= total) return;
    int j = idx % W;
    int t = idx / W;
    int i = t % H; t /= H;
    int ch = t % 3;
    int b  = t / 3;

    const size_t cb = ((size_t)(b * (H + 2) + i + 1) * (W + 2) + j + 1) * 64;
    const float* base1 = t1 + (size_t)(b * 3 + ch) * H * W;
    const float* base2 = t2 + (size_t)(b * 3 + ch) * H * W;

    float acc = 0.f;
    for (int u = 0; u < KSZ; ++u) {
        int ih = i + u - KHALF;
        ih = ih < 0 ? 0 : (ih >= H ? H - 1 : ih);
        const float* r1 = base1 + (size_t)ih * W;
        const float* r2 = base2 + (size_t)ih * W;
        float h1 = 0.f, h2 = 0.f;
        for (int v = 0; v < KSZ; ++v) {
            int iw = j + v - KHALF;
            iw = iw < 0 ? 0 : (iw >= W ? W - 1 : iw);
            h1 += r1[iw] * (float)Hk1[cb + v];
            h2 += r2[iw] * (float)Hk2[cb + v];
        }
        acc += (float)V1[cb + u] * h1 + (float)V2[cb + u] * h2;
    }
    out[idx] = acc;
}

// ---------------------------------------------------------------------------
// Host orchestration
// ---------------------------------------------------------------------------
static void run_conv(hipStream_t st, const _Float16* in,
                     const float* wsrc, const float* bsrc,
                     _Float16* wbuf, float* bbuf, _Float16* out,
                     int Bn, int H, int W, int cin, int cout, int relu)
{
    const int cinP  = pad32i(cin);
    const int coutP = pad32i(cout);
    repack_w_kernel<<<icdiv(9 * cinP * coutP, 256), 256, 0, st>>>(
        wsrc, wbuf, cin, cout, cinP, coutP);
    repack_b_kernel<<<icdiv(coutP, 64), 64, 0, st>>>(bsrc, bbuf, cout, coutP);
    const int nPix = Bn * H * W;                    // always a multiple of 64
    dim3 g(nPix / 64, coutP / 32);
    conv3x3_wmma_kernel<<<g, 128, 0, st>>>(in, wbuf, bbuf, out,
                                           H, W, cinP, coutP, relu);
}

extern "C" void kernel_launch(void* const* d_in, const int* in_sizes, int n_in,
                              void* d_out, int out_size, void* d_ws, size_t ws_size,
                              hipStream_t stream)
{
    (void)in_sizes; (void)n_in; (void)out_size; (void)ws_size;
    const int Bn = 2, H0 = 256, W0 = 256;

    const float* T1 = (const float*)d_in[0];
    const float* T2 = (const float*)d_in[1];
    // params flattened in insertion order: basics(9x3x(w,b)), ups(4x(w,b)),
    // subnets(4x4x(w,b))
    auto P  = [&](int i) { return (const float*)d_in[i]; };
    auto BW = [&](int blk, int j) { return 2 + (blk * 3 + j) * 2; };
    auto UW = [&](int i)          { return 56 + 2 * i;          };
    auto SW = [&](int s, int j)   { return 64 + (s * 4 + j) * 2; };

    // bump allocator over d_ws
    char* ws = (char*)d_ws;
    size_t off = 0;
    auto A16 = [&](size_t n) -> _Float16* {
        off = (off + 255) & ~(size_t)255;
        _Float16* p = (_Float16*)(ws + off); off += n * sizeof(_Float16); return p;
    };
    auto A32 = [&](size_t n) -> float* {
        off = (off + 255) & ~(size_t)255;
        float* p = (float*)(ws + off); off += n * sizeof(float); return p;
    };

    _Float16* wbuf = A16((size_t)9 * 512 * 512);    // reused weight scratch
    float*    bbuf = A32(512);

    // allocate a halo tensor and zero its border
    auto tensor = [&](int H, int W, int C) -> _Float16* {
        _Float16* p = A16((size_t)Bn * (H + 2) * (W + 2) * C);
        int npx = Bn * (H + 2) * (W + 2);
        zero_halo_kernel<<<icdiv(npx, 256), 256, 0, stream>>>(p, Bn, H, W, C);
        return p;
    };

    auto conv = [&](const _Float16* in, int widx, _Float16* out,
                    int H, int W, int cin, int cout, int relu) {
        run_conv(stream, in, P(widx), P(widx + 1), wbuf, bbuf, out,
                 Bn, H, W, cin, cout, relu);
    };
    auto basic = [&](const _Float16* in, int H, int W,
                     int cin, int cout, int blk) -> _Float16* {
        _Float16* o1 = tensor(H, W, pad32i(cout));
        _Float16* o2 = tensor(H, W, pad32i(cout));
        _Float16* o3 = tensor(H, W, pad32i(cout));
        conv(in, BW(blk, 0), o1, H, W, cin,  cout, 1);
        conv(o1, BW(blk, 1), o2, H, W, cout, cout, 1);
        conv(o2, BW(blk, 2), o3, H, W, cout, cout, 1);
        return o3;
    };
    auto pool = [&](const _Float16* in, int H, int W, int C) -> _Float16* {
        _Float16* o = tensor(H / 2, W / 2, C);
        int n = Bn * (H / 2) * (W / 2) * C;
        avgpool_kernel<<<icdiv(n, 256), 256, 0, stream>>>(in, o, Bn, H, W, C);
        return o;
    };
    auto up2f = [&](const _Float16* in, int H, int W, int C) -> _Float16* {
        _Float16* o = tensor(2 * H, 2 * W, C);
        int n = Bn * 2 * H * 2 * W * C;
        up2_kernel<<<icdiv(n, 256), 256, 0, stream>>>(in, o, Bn, H, W, C);
        return o;
    };
    auto addf = [&](const _Float16* a, const _Float16* b,
                    int H, int W, int C) -> _Float16* {
        _Float16* o = A16((size_t)Bn * (H + 2) * (W + 2) * C);
        int n = Bn * (H + 2) * (W + 2) * C;   // full extent: halos are zero
        add_kernel<<<icdiv(n, 256), 256, 0, stream>>>(a, b, o, n);
        return o;
    };
    auto upblock = [&](const _Float16* x, int H, int W, int C, int ui) -> _Float16* {
        _Float16* u = up2f(x, H, W, C);
        _Float16* o = tensor(2 * H, 2 * W, C);
        conv(u, UW(ui), o, 2 * H, 2 * W, C, C, 1);
        return o;
    };

    // ---- encoder ----
    const int n0 = Bn * H0 * W0;
    _Float16* xin = tensor(256, 256, 32);
    pack_kernel<<<icdiv(n0, 256), 256, 0, stream>>>(T1, T2, xin, Bn, H0, W0);

    _Float16* c1 = basic(xin, 256, 256,   6,  32, 0);
    _Float16* c2 = basic(pool(c1, 256, 256,  32), 128, 128,  32,  64, 1);
    _Float16* c3 = basic(pool(c2, 128, 128,  64),  64,  64,  64, 128, 2);
    _Float16* c4 = basic(pool(c3,  64,  64, 128),  32,  32, 128, 256, 3);
    _Float16* c5 = basic(pool(c4,  32,  32, 256),  16,  16, 256, 512, 4);
    _Float16* c6 = basic(pool(c5,  16,  16, 512),   8,   8, 512, 512, 5);

    // ---- decoder ----
    _Float16* s6  = addf(upblock(c6,  8,  8, 512, 0), c5, 16, 16, 512);
    _Float16* c7  = basic(s6, 16, 16, 512, 256, 6);
    _Float16* s7  = addf(upblock(c7, 16, 16, 256, 1), c4, 32, 32, 256);
    _Float16* c8  = basic(s7, 32, 32, 256, 128, 7);
    _Float16* s8  = addf(upblock(c8, 32, 32, 128, 2), c3, 64, 64, 128);
    _Float16* c9  = basic(s8, 64, 64, 128, 64, 8);
    _Float16* c10 = upblock(c9, 64, 64, 64, 3);
    _Float16* feat = addf(c10, c2, 128, 128, 64);

    // ---- kernel-prediction subnets (V1, V2, Hk1, Hk2) ----
    _Float16* sa = tensor(128, 128, 64);
    _Float16* sb = tensor(128, 128, 64);
    _Float16* sc = tensor(128, 128, 64);
    _Float16* Vt[4];
    for (int s = 0; s < 4; ++s) {
        conv(feat, SW(s, 0), sa, 128, 128, 64, 64, 1);
        conv(sa,   SW(s, 1), sb, 128, 128, 64, 64, 1);
        conv(sb,   SW(s, 2), sc, 128, 128, 64, KSZ, 1);
        _Float16* sup = up2f(sc, 128, 128, 64);          // -> 256^2 x 64
        Vt[s] = tensor(256, 256, 64);
        conv(sup, SW(s, 3), Vt[s], 256, 256, KSZ, KSZ, 0);  // final: no ReLU
    }

    // ---- adaptive separable conv readout ----
    const int total = Bn * 3 * H0 * W0;
    sepconv_kernel<<<icdiv(total, 256), 256, 0, stream>>>(
        T1, T2, Vt[0], Vt[1], Vt[2], Vt[3], (float*)d_out, Bn, H0, W0);
}
